// convolution_56410100466092
// MI455X (gfx1250) — matrix operand order repaired and checked by reading, not compiled, not run on hardware
//
#include <hip/hip_runtime.h>

typedef __attribute__((ext_vector_type(16))) _Float16 v16h;
typedef __attribute__((ext_vector_type(8)))  float    v8f;

__device__ __forceinline__ void atom_add_f32(float* p, float v) {
  // relaxed agent-scope fp32 add, no return -> global_atomic_add_f32
  __hip_atomic_fetch_add(p, v, __ATOMIC_RELAXED, __HIP_MEMORY_SCOPE_AGENT);
}

// ---------------- zero the aggregation buffer (needed every launch: atomics) ----
__global__ void k_zero(float4* __restrict__ p, long n4) {
  long i = (long)blockIdx.x * blockDim.x + threadIdx.x;
  long stride = (long)gridDim.x * blockDim.x;
  float4 z = make_float4(0.f, 0.f, 0.f, 0.f);
  for (; i < n4; i += stride) p[i] = z;
}

// ---------------- pack MLP weights into f16 WMMA B-fragment layout ----------
// halves index = base + f*512 + lane*16 + e
//   f = tile*2 + chunk ; value = W[K*ncols + col]
//   K = chunk*32 + (e>=8 ? 16 : 0) + (lane>=16 ? 8 : 0) + (e&7) ; col = tile*16 + (lane&15)
__global__ void k_pack_w(const float* __restrict__ W1, const float* __restrict__ W2,
                         _Float16* __restrict__ out) {
  int tid = blockIdx.x * blockDim.x + threadIdx.x;
  for (int idx = tid; idx < 4096 + 5120; idx += blockDim.x * gridDim.x) {
    const float* W; int ncols; int local;
    if (idx < 4096) { W = W1; ncols = 64; local = idx; }
    else            { W = W2; ncols = 80; local = idx - 4096; }
    int f    = local >> 9;
    int lane = (local >> 4) & 31;
    int e    = local & 15;
    int tile = f >> 1, chunk = f & 1;
    int K    = chunk * 32 + ((e >> 3) << 4) + ((lane >> 4) << 3) + (e & 7);
    int col  = tile * 16 + (lane & 15);
    out[idx] = (_Float16)W[K * ncols + col];
  }
}

// ---------------- per-node self-interaction s = fctp(x, W_si), sk = fctp(x, W_sc) --
// layout of s/sk rows (64 floats): [u]=s0[u] for u<16 ; [16+3u+i]=s1[u][i]
__global__ void k_node_stage(const float* __restrict__ x, const float* __restrict__ attr,
                             const float* __restrict__ Wsi0, const float* __restrict__ Wsi1,
                             const float* __restrict__ Wsc0, const float* __restrict__ Wsc1,
                             float* __restrict__ s, float* __restrict__ sk, int N) {
  int gid = blockIdx.x * blockDim.x + threadIdx.x;
  if (gid >= N * 16) return;
  int n = gid >> 4, u = gid & 15;
  float a = attr[n] * 0.25f;               // attr / sqrt(16)
  const float* xr = x + (long)n * 64;
  float s0 = 0.f, sk0 = 0.f, s1[3] = {0,0,0}, sk1[3] = {0,0,0};
#pragma unroll
  for (int v = 0; v < 16; v++) {
    float xv = xr[v];
    s0  += xv * Wsi0[u*16+v];
    sk0 += xv * Wsc0[u*16+v];
    float wi1 = Wsi1[u*16+v], wc1 = Wsc1[u*16+v];
#pragma unroll
    for (int i = 0; i < 3; i++) {
      float x1 = xr[16 + 3*v + i];
      s1[i]  += x1 * wi1;
      sk1[i] += x1 * wc1;
    }
  }
  float* sr  = s  + (long)n * 64;
  float* skr = sk + (long)n * 64;
  sr[u] = s0 * a;  skr[u] = sk0 * a;
#pragma unroll
  for (int i = 0; i < 3; i++) { sr[16+3*u+i] = s1[i]*a; skr[16+3*u+i] = sk1[i]*a; }
}

// ---------------- fused edge kernel: MLP (WMMA) + gather + messages + scatter ----
__global__ __launch_bounds__(256) void k_edge(
    const float* __restrict__ dist,
    const int*   __restrict__ esrc, const int* __restrict__ edst,
    const float* __restrict__ eattr,
    const _Float16* __restrict__ wF,
    const float* __restrict__ sbuf,
    float* __restrict__ agg, int E)
{
  __shared__ float lh[8][16 * 64];          // per-wave h staging, 32 KB/block
  const int lane = threadIdx.x & 31;
  const int wid  = threadIdx.x >> 5;
  const long tile = (long)blockIdx.x * 8 + wid;
  const int eBase = (int)(tile * 16);
  if (eBase >= E) return;                   // no block-wide barriers used below
  const int hi = lane >> 4;
  const int n0 = lane & 15;
  const int kb = hi * 8;

  // ---- A fragments: 16 dist rows, f32 -> f16 (ISA 16-bit 16x32 A layout) ----
  int rowA = eBase + n0; if (rowA >= E) rowA = E - 1;
  const float* dr = dist + (long)rowA * 64;
  v16h aF[2];
#pragma unroll
  for (int c = 0; c < 2; c++) {
    const float4* r0 = (const float4*)(dr + c*32 + kb);
    const float4* r1 = (const float4*)(dr + c*32 + 16 + kb);
    float4 p0 = r0[0], p1 = r0[1], q0 = r1[0], q1 = r1[1];
    v16h av;
    av[0]=(_Float16)p0.x; av[1]=(_Float16)p0.y; av[2]=(_Float16)p0.z; av[3]=(_Float16)p0.w;
    av[4]=(_Float16)p1.x; av[5]=(_Float16)p1.y; av[6]=(_Float16)p1.z; av[7]=(_Float16)p1.w;
    av[8]=(_Float16)q0.x; av[9]=(_Float16)q0.y; av[10]=(_Float16)q0.z; av[11]=(_Float16)q0.w;
    av[12]=(_Float16)q1.x; av[13]=(_Float16)q1.y; av[14]=(_Float16)q1.z; av[15]=(_Float16)q1.w;
    aF[c] = av;
  }

  // ---- GEMM1: h(16x64) = dist_tile @ W_mlp1 ----
  v8f h[4];
#pragma unroll
  for (int t = 0; t < 4; t++) {
    v8f c = {};
    v16h b0 = *(const v16h*)(wF + ((t*2+0)*512 + lane*16));
    v16h b1 = *(const v16h*)(wF + ((t*2+1)*512 + lane*16));
    c = __builtin_amdgcn_wmma_f32_16x16x32_f16(false, aF[0], false, b0, (short)0, c, false, false);
    c = __builtin_amdgcn_wmma_f32_16x16x32_f16(false, aF[1], false, b1, (short)0, c, false, false);
    h[t] = c;
  }

  // ---- silu(h/8), stage to this wave's LDS slab ----
  float* myh = &lh[wid][0];
#pragma unroll
  for (int t = 0; t < 4; t++) {
#pragma unroll
    for (int r = 0; r < 8; r++) {
      float xv = h[t][r] * 0.125f;                          // / sqrt(64)
      xv = xv * __builtin_amdgcn_rcpf(1.0f + __expf(-xv));  // silu via v_rcp_f32
      myh[(hi*8 + r)*64 + t*16 + n0] = xv;                  // row-major [M][64]
    }
  }
  asm volatile("s_wait_dscnt 0x0" ::: "memory");   // cross-lane DS RAW inside wave

  // ---- reload h in A-fragment layout ----
  const float* hrow = myh + n0 * 64;
  v16h hA[2];
#pragma unroll
  for (int c = 0; c < 2; c++) {
    const float* r0 = hrow + c*32 + kb;
    const float* r1 = hrow + c*32 + 16 + kb;
    v16h av;
#pragma unroll
    for (int j = 0; j < 8; j++) { av[j] = (_Float16)r0[j]; av[8+j] = (_Float16)r1[j]; }
    hA[c] = av;
  }

  // ---- GEMM2: w(16x80) = h @ W_mlp2 ; tile t == w_t ----
  v8f wv[5];
#pragma unroll
  for (int t = 0; t < 5; t++) {
    v8f c = {};
    v16h b0 = *(const v16h*)(wF + (4096 + (t*2+0)*512 + lane*16));
    v16h b1 = *(const v16h*)(wF + (4096 + (t*2+1)*512 + lane*16));
    c = __builtin_amdgcn_wmma_f32_16x16x32_f16(false, hA[0], false, b0, (short)0, c, false, false);
    c = __builtin_amdgcn_wmma_f32_16x16x32_f16(false, hA[1], false, b1, (short)0, c, false, false);
    wv[t] = c;
  }
  const float S  = 0.03125f;                 // (1/sqrt(64)) * (1/sqrt(16 neighbors))
  const float S3 = S * 0.57735026919f;       // extra 1/sqrt(3)
  const float S2 = S * 0.70710678119f;       // extra 1/sqrt(2)

  // ---- per-half-wave edge indices: 8 contiguous rows -> vector loads ----
  int srcs[8], dsts[8];
  if (eBase + 16 <= E) {
    int4 s0v = *(const int4*)(esrc + eBase + hi*8);
    int4 s1v = *(const int4*)(esrc + eBase + hi*8 + 4);
    int4 d0v = *(const int4*)(edst + eBase + hi*8);
    int4 d1v = *(const int4*)(edst + eBase + hi*8 + 4);
    srcs[0]=s0v.x; srcs[1]=s0v.y; srcs[2]=s0v.z; srcs[3]=s0v.w;
    srcs[4]=s1v.x; srcs[5]=s1v.y; srcs[6]=s1v.z; srcs[7]=s1v.w;
    dsts[0]=d0v.x; dsts[1]=d0v.y; dsts[2]=d0v.z; dsts[3]=d0v.w;
    dsts[4]=d1v.x; dsts[5]=d1v.y; dsts[6]=d1v.z; dsts[7]=d1v.w;
  } else {
#pragma unroll
    for (int r = 0; r < 8; r++) {
      int e = eBase + hi*8 + r;
      srcs[r] = (e < E) ? esrc[e] : 0;
      dsts[r] = (e < E) ? edst[e] : 0;
    }
  }

  // ---- messages + scatter: lane owns channel u=n0 of rows M = hi*8+r ----
  const int u = n0;
#pragma unroll
  for (int r = 0; r < 8; r++) {
    int e = eBase + hi*8 + r;
    if (e >= E) continue;
    float w0 = wv[0][r]*S, w1 = wv[1][r]*S, w2 = wv[2][r]*S;
    float w3 = wv[3][r]*S3, w4 = wv[4][r]*S2;
    int dst = dsts[r], src = srcs[r];
    float4 ya = ((const float4*)eattr)[e];
    float y0 = ya.x, y1x = ya.y, y1y = ya.z, y1z = ya.w;
    const float* sp = sbuf + (long)dst * 64;
    float g0  = sp[u];
    float g1x = sp[16+3*u+0], g1y = sp[16+3*u+1], g1z = sp[16+3*u+2];
    float m0a = w0 * g0 * y0;
    float m0b = w3 * (g1x*y1x + g1y*y1y + g1z*y1z);
    float wg = w1 * g0, wy = w2 * y0;
    float cx = g1y*y1z - g1z*y1y;
    float cy = g1z*y1x - g1x*y1z;
    float cz = g1x*y1y - g1y*y1x;
    float* an = agg + (long)src * 176;
    atom_add_f32(an + u,            m0a);
    atom_add_f32(an + 16 + u,       m0b);
    atom_add_f32(an + 32  + 3*u + 0, wg*y1x);
    atom_add_f32(an + 32  + 3*u + 1, wg*y1y);
    atom_add_f32(an + 32  + 3*u + 2, wg*y1z);
    atom_add_f32(an + 80  + 3*u + 0, wy*g1x);
    atom_add_f32(an + 80  + 3*u + 1, wy*g1y);
    atom_add_f32(an + 80  + 3*u + 2, wy*g1z);
    atom_add_f32(an + 128 + 3*u + 0, w4*cx);
    atom_add_f32(an + 128 + 3*u + 1, w4*cy);
    atom_add_f32(an + 128 + 3*u + 2, w4*cz);
  }
}

// ---------------- per-node finalize: fctp(agg, W_lo)*alpha + skip -> out ----
__global__ void k_finalize(const float* __restrict__ agg, const float* __restrict__ sk,
                           const float* __restrict__ attr,
                           const float* __restrict__ Wlo0, const float* __restrict__ Wlo1,
                           const float* __restrict__ Walpha,
                           float* __restrict__ out, int N)
{
  int gid = blockIdx.x * blockDim.x + threadIdx.x;
  if (gid >= N * 16) return;
  int n = gid >> 4, u = gid & 15;
  float a = attr[n];
  const float* ag = agg + (long)n * 176;
  float o0 = 0.f, alpha = 0.f;
#pragma unroll
  for (int v = 0; v < 32; v++) {
    float av = ag[v];
    o0    += av * Wlo0[u*32+v];
    alpha += av * Walpha[v];
  }
  const float inv32 = 0.17677669529663687f;   // 1/sqrt(32)
  const float inv48 = 0.14433756729740643f;   // 1/sqrt(48)
  o0    *= a * inv32;
  alpha *= a * inv32;
  float o1[3] = {0,0,0};
#pragma unroll
  for (int v = 0; v < 48; v++) {
    float wl = Wlo1[u*48+v];
    o1[0] += ag[32+3*v+0] * wl;
    o1[1] += ag[32+3*v+1] * wl;
    o1[2] += ag[32+3*v+2] * wl;
  }
  const float* skr = sk + (long)n * 64;
  out[(long)n*64 + u] = o0 * alpha + skr[u];
#pragma unroll
  for (int i = 0; i < 3; i++)
    out[(long)n*64 + 16 + 3*u + i] = (o1[i]*a*inv48) * alpha + skr[16+3*u+i];
}

extern "C" void kernel_launch(void* const* d_in, const int* in_sizes, int n_in,
                              void* d_out, int out_size, void* d_ws, size_t ws_size,
                              hipStream_t stream)
{
  (void)n_in; (void)out_size; (void)ws_size;
  const float* node_input = (const float*)d_in[0];
  const float* node_attr  = (const float*)d_in[1];
  const int*   edge_src   = (const int*)  d_in[2];
  const int*   edge_dst   = (const int*)  d_in[3];
  const float* edge_attr  = (const float*)d_in[4];
  const float* dist_emb   = (const float*)d_in[5];
  const float* W_si0  = (const float*)d_in[6];
  const float* W_si1  = (const float*)d_in[7];
  const float* W_sc0  = (const float*)d_in[8];
  const float* W_sc1  = (const float*)d_in[9];
  const float* W_mlp1 = (const float*)d_in[10];
  const float* W_mlp2 = (const float*)d_in[11];
  const float* W_lo0  = (const float*)d_in[12];
  const float* W_lo1  = (const float*)d_in[13];
  const float* W_alph = (const float*)d_in[14];
  int N = in_sizes[0] / 64;
  int E = in_sizes[2];

  char* ws = (char*)d_ws;
  _Float16* wF   = (_Float16*)ws;                    // 9216 halves (18.4 KB)
  float*    sbuf = (float*)(ws + 32768);             // N*64
  float*    skbf = sbuf + (size_t)N * 64;            // N*64
  float*    agg  = skbf + (size_t)N * 64;            // N*176

  hipLaunchKernelGGL(k_pack_w, dim3(9), dim3(256), 0, stream, W_mlp1, W_mlp2, wF);
  hipLaunchKernelGGL(k_zero, dim3(2048), dim3(256), 0, stream,
                     (float4*)agg, (long)N * 44);
  hipLaunchKernelGGL(k_node_stage, dim3((N*16 + 255)/256), dim3(256), 0, stream,
                     node_input, node_attr, W_si0, W_si1, W_sc0, W_sc1, sbuf, skbf, N);
  long tiles = (E + 15) / 16;
  int eblocks = (int)((tiles + 7) / 8);
  hipLaunchKernelGGL(k_edge, dim3(eblocks), dim3(256), 0, stream,
                     dist_emb, edge_src, edge_dst, edge_attr, wF, sbuf, agg, E);
  hipLaunchKernelGGL(k_finalize, dim3((N*16 + 255)/256), dim3(256), 0, stream,
                     agg, skbf, node_attr, W_lo0, W_lo1, W_alph, (float*)d_out, N);
}
